// SelfAttention_3590592660075
// MI455X (gfx1250) — matrix operand order repaired and checked by reading, not compiled, not run on hardware
//
#include <hip/hip_runtime.h>
#include <hip/hip_bf16.h>

typedef __bf16 bf16_t;
typedef __attribute__((ext_vector_type(16))) __bf16 v16bf;
typedef __attribute__((ext_vector_type(8)))  __bf16 v8bf;
typedef __attribute__((ext_vector_type(8)))  float  v8f;

#define BB 4
#define SS 2048
#define DD 1024
#define NROWS (BB*SS)   // 8192 total (b,s) rows

// ---------------------------------------------------------------------------
// Async global -> LDS copy of one 16B chunk (GLOBAL_LOAD_ASYNC_TO_LDS_B128,
// tracked by ASYNCcnt).  Generic pointer to __shared__ = aperture|lds_offset,
// so the low 32 bits are exactly the LDS byte address the instruction wants.
// ---------------------------------------------------------------------------
__device__ __forceinline__ void async_g2l_b128(const bf16_t* g, bf16_t* l) {
  unsigned lofs = (unsigned)(size_t)l;
  asm volatile("global_load_async_to_lds_b128 %0, %1, off"
               :: "v"(lofs), "v"(g) : "memory");
}
__device__ __forceinline__ void wait_async0() {
  asm volatile("s_wait_asynccnt 0x0" ::: "memory");
}

// ---------------------------------------------------------------------------
// WMMA fragment loaders (bf16 16x16x32, wave32 layouts per CDNA5 ISA 7.12.2)
// A: lanes 0-15 row M=lane,   elems 0-7 -> K=kk+0..7, 8-15 -> K=kk+16..23
//    lanes16-31 row M=lane-16, elems 0-7 -> K=kk+8..15, 8-15 -> K=kk+24..31
// B from Bt (N x K row-major): same K order as A (half h: K=kk+8h+0..7, kk+8h+16..23)
// ---------------------------------------------------------------------------
__device__ __forceinline__ v16bf frag_a(const bf16_t* A, int lda, int r0, int kk,
                                        int lane) {
  int row  = r0 + (lane & 15);
  int koff = kk + ((lane >> 4) << 3);
  const bf16_t* p = A + (size_t)row * lda + koff;
  v8bf lo = *(const v8bf*)(p);
  v8bf hi = *(const v8bf*)(p + 16);
  v16bf r;
#pragma unroll
  for (int i = 0; i < 8; ++i) { r[i] = lo[i]; r[i + 8] = hi[i]; }
  return r;
}

__device__ __forceinline__ v16bf frag_b(const bf16_t* Bt, int ldb, int n0, int kk,
                                        int lane) {
  int col  = n0 + (lane & 15);
  int koff = kk + ((lane >> 4) << 3);
  const bf16_t* p = Bt + (size_t)col * ldb + koff;
  v8bf lo = *(const v8bf*)(p);
  v8bf hi = *(const v8bf*)(p + 16);
  v16bf r;
#pragma unroll
  for (int i = 0; i < 8; ++i) { r[i] = lo[i]; r[i + 8] = hi[i]; }
  return r;
}

__device__ __forceinline__ v8f wmma_bf16(v16bf a, v16bf b, v8f c) {
  return __builtin_amdgcn_wmma_f32_16x16x32_bf16(false, a, false, b, (short)0, c,
                                                 false, false);
}

// ---------------------------------------------------------------------------
// Kernel 0: fp32 -> bf16 conversion
// ---------------------------------------------------------------------------
__global__ void f32_to_bf16_kernel(const float* __restrict__ src,
                                   bf16_t* __restrict__ dst, int n) {
  int i = blockIdx.x * blockDim.x + threadIdx.x;
  if (i < n) dst[i] = (bf16_t)src[i];
}

// ---------------------------------------------------------------------------
// Kernel 1: fused projection GEMM + bias + LayerNorm.  M-tile = 32 rows.
//   blockIdx.x: 32-row tile; blockIdx.y: 0=q,1=k,2=v
//   A strip (32 x 512 bf16 = 32KB) staged to LDS per K-phase via async copies.
//   q,k row-major bf16; v transposed [b][d][s] bf16.
// ---------------------------------------------------------------------------
__global__ __launch_bounds__(512)
void qkv_ln_kernel(const bf16_t* __restrict__ xb, const bf16_t* __restrict__ wb,
                   const float* __restrict__ bq, const float* __restrict__ bk,
                   const float* __restrict__ bv, const float* __restrict__ gamma,
                   const float* __restrict__ beta, bf16_t* __restrict__ qb,
                   bf16_t* __restrict__ kb, bf16_t* __restrict__ vtb) {
  __shared__ alignas(16) bf16_t tileA[32 * 512];   // 32 KB
  __shared__ float ln_sum[32];
  __shared__ float ln_sq[32];

  const int tid  = threadIdx.x;
  const int w    = tid >> 5;
  const int lane = tid & 31;
  const int hi   = lane >> 4;
  const int m0   = blockIdx.x << 5;
  const int proj = blockIdx.y;

  if (tid < 32) { ln_sum[tid] = 0.0f; ln_sq[tid] = 0.0f; }

  const bf16_t* W = wb + (size_t)proj * DD * DD;
  const float* biasp = (proj == 0) ? bq : (proj == 1) ? bk : bv;

  v8f acc[2][4];
#pragma unroll
  for (int rt = 0; rt < 2; ++rt)
#pragma unroll
    for (int t = 0; t < 4; ++t) acc[rt][t] = (v8f){};

  for (int ph = 0; ph < 2; ++ph) {
    __syncthreads();  // previous phase fully consumed (and ln init visible)
    // stage 32 rows x 512 cols of x into LDS: 2048 chunks of 16B, 4/thread
#pragma unroll
    for (int j = 0; j < 4; ++j) {
      int c   = tid + (j << 9);
      int row = c >> 6;              // 64 chunks per row
      int cc  = (c & 63) << 3;       // element offset within 512
      async_g2l_b128(xb + (size_t)(m0 + row) * DD + (ph << 9) + cc,
                     &tileA[(row << 9) + cc]);
    }
    wait_async0();
    __syncthreads();

    for (int kk2 = 0; kk2 < 512; kk2 += 32) {
      int kk = (ph << 9) + kk2;
      v16bf a0 = frag_a(tileA, 512, 0, kk2, lane);
      v16bf a1 = frag_a(tileA, 512, 16, kk2, lane);
      __builtin_prefetch(W + (size_t)((w << 6) + (lane & 15)) * DD + kk + 256, 0, 1);
#pragma unroll
      for (int t = 0; t < 4; ++t) {
        int n0 = ((w << 2) + t) << 4;
        v16bf b = frag_b(W, DD, n0, kk, lane);
        acc[0][t] = wmma_bf16(a0, b, acc[0][t]);
        acc[1][t] = wmma_bf16(a1, b, acc[1][t]);
      }
    }
  }

  // bias
#pragma unroll
  for (int t = 0; t < 4; ++t) {
    int col = (((w << 2) + t) << 4) + (lane & 15);
    float bval = biasp[col];
#pragma unroll
    for (int rt = 0; rt < 2; ++rt)
#pragma unroll
      for (int r = 0; r < 8; ++r) acc[rt][t][r] += bval;
  }

  __syncthreads();

  // LayerNorm statistics per row (sum / sumsq across 1024 columns)
#pragma unroll
  for (int rt = 0; rt < 2; ++rt) {
#pragma unroll
    for (int r = 0; r < 8; ++r) {
      float s = 0.0f, q = 0.0f;
#pragma unroll
      for (int t = 0; t < 4; ++t) { float v = acc[rt][t][r]; s += v; q += v * v; }
#pragma unroll
      for (int off = 1; off < 16; off <<= 1) {
        s += __shfl_xor(s, off, 32);
        q += __shfl_xor(q, off, 32);
      }
      if ((lane & 15) == 0) {
        int m = (rt << 4) + r + (hi << 3);
        atomicAdd(&ln_sum[m], s);
        atomicAdd(&ln_sq[m], q);
      }
    }
  }
  __syncthreads();

  // normalize + affine + store
#pragma unroll
  for (int t = 0; t < 4; ++t) {
    int col = (((w << 2) + t) << 4) + (lane & 15);
    float g = gamma[col], bta = beta[col];
#pragma unroll
    for (int rt = 0; rt < 2; ++rt) {
#pragma unroll
      for (int r = 0; r < 8; ++r) {
        int m = (rt << 4) + r + (hi << 3);
        float mu  = ln_sum[m] * (1.0f / DD);
        float var = ln_sq[m] * (1.0f / DD) - mu * mu;
        float rs  = rsqrtf(var + 1e-5f);
        float val = (acc[rt][t][r] - mu) * rs * g + bta;
        int grow = m0 + m;
        if (proj == 0) {
          qb[(size_t)grow * DD + col] = (bf16_t)val;
        } else if (proj == 1) {
          kb[(size_t)grow * DD + col] = (bf16_t)val;
        } else {
          int bidx = grow >> 11;            // / SS
          int srow = grow & (SS - 1);       // % SS
          vtb[((size_t)bidx * DD + col) * SS + srow] = (bf16_t)val;
        }
      }
    }
  }
}

// ---------------------------------------------------------------------------
// Kernel 2: scores + softmax.  Block: 16 query rows x all 2048 keys (1 batch).
//   Q strip (16 x 1024 = 32KB) staged to LDS once via async copies.
// ---------------------------------------------------------------------------
__global__ __launch_bounds__(512)
void scores_softmax_kernel(const bf16_t* __restrict__ qb,
                           const bf16_t* __restrict__ kb,
                           bf16_t* __restrict__ Pb) {
  __shared__ alignas(16) bf16_t tileQ[16 * DD];   // 32 KB
  __shared__ float red[16][16];
  __shared__ float rowmax_s[16];
  __shared__ float rowsum_s[16];

  const int tid  = threadIdx.x;
  const int w    = tid >> 5;
  const int lane = tid & 31;
  const int hi   = lane >> 4;
  const int m0   = blockIdx.x << 4;
  const int batch = m0 >> 11;
  const int krow0 = batch << 11;

  // stage Q strip: 2048 chunks of 16B, 4 per thread (rows are contiguous)
#pragma unroll
  for (int j = 0; j < 4; ++j) {
    int c = (tid + (j << 9)) << 3;   // element index in the flat 16x1024 strip
    async_g2l_b128(qb + (size_t)m0 * DD + c, &tileQ[c]);
  }
  wait_async0();
  __syncthreads();

  v8f acc[8];
#pragma unroll
  for (int t = 0; t < 8; ++t) acc[t] = (v8f){};

  for (int kk = 0; kk < DD; kk += 32) {
    v16bf a = frag_a(tileQ, DD, 0, kk, lane);
    __builtin_prefetch(kb + (size_t)(krow0 + (w << 7) + (lane & 15)) * DD + kk + 256,
                       0, 1);
#pragma unroll
    for (int t = 0; t < 8; ++t) {
      int n0 = krow0 + (((w << 3) + t) << 4);
      v16bf b = frag_b(kb, DD, n0, kk, lane);
      acc[t] = wmma_bf16(a, b, acc[t]);
    }
  }

  // row max
#pragma unroll
  for (int r = 0; r < 8; ++r) {
    float mx = -3.4e38f;
#pragma unroll
    for (int t = 0; t < 8; ++t) mx = fmaxf(mx, acc[t][r]);
#pragma unroll
    for (int off = 1; off < 16; off <<= 1) mx = fmaxf(mx, __shfl_xor(mx, off, 32));
    if ((lane & 15) == 0) red[w][r + (hi << 3)] = mx;
  }
  __syncthreads();
  if (tid < 16) {
    float mx = red[0][tid];
#pragma unroll
    for (int ww = 1; ww < 16; ++ww) mx = fmaxf(mx, red[ww][tid]);
    rowmax_s[tid] = mx;
  }
  __syncthreads();

  // exp + row sum
#pragma unroll
  for (int r = 0; r < 8; ++r) {
    int m = r + (hi << 3);
    float rm = rowmax_s[m];
    float s = 0.0f;
#pragma unroll
    for (int t = 0; t < 8; ++t) {
      float p = __expf(acc[t][r] - rm);
      acc[t][r] = p;
      s += p;
    }
#pragma unroll
    for (int off = 1; off < 16; off <<= 1) s += __shfl_xor(s, off, 32);
    if ((lane & 15) == 0) red[w][m] = s;
  }
  __syncthreads();
  if (tid < 16) {
    float s = red[0][tid];
#pragma unroll
    for (int ww = 1; ww < 16; ++ww) s += red[ww][tid];
    rowsum_s[tid] = s;
  }
  __syncthreads();

  // normalize + store bf16 probabilities
#pragma unroll
  for (int t = 0; t < 8; ++t) {
    int key = (((w << 3) + t) << 4) + (lane & 15);
#pragma unroll
    for (int r = 0; r < 8; ++r) {
      int m = r + (hi << 3);
      float val = acc[t][r] * (1.0f / rowsum_s[m]);
      Pb[(size_t)(m0 + m) * SS + key] = (bf16_t)val;
    }
  }
}

// ---------------------------------------------------------------------------
// Kernel 3: out = P @ V with transposed V (vt[b][d][s]).  M-tile = 32 rows.
//   P strip staged to LDS in 4 phases of 32 x 512 keys (32KB each).
// ---------------------------------------------------------------------------
__global__ __launch_bounds__(512)
void pv_kernel(const bf16_t* __restrict__ Pb, const bf16_t* __restrict__ vtb,
               float* __restrict__ out) {
  __shared__ alignas(16) bf16_t tileP[32 * 512];   // 32 KB

  const int tid  = threadIdx.x;
  const int w    = tid >> 5;
  const int lane = tid & 31;
  const int hi   = lane >> 4;
  const int m0   = blockIdx.x << 5;
  const int batch = m0 >> 11;
  const bf16_t* vt = vtb + (size_t)batch * DD * SS;

  v8f acc[2][4];
#pragma unroll
  for (int rt = 0; rt < 2; ++rt)
#pragma unroll
    for (int t = 0; t < 4; ++t) acc[rt][t] = (v8f){};

  for (int ph = 0; ph < 4; ++ph) {
    __syncthreads();
#pragma unroll
    for (int j = 0; j < 4; ++j) {
      int c   = tid + (j << 9);
      int row = c >> 6;
      int cc  = (c & 63) << 3;
      async_g2l_b128(Pb + (size_t)(m0 + row) * SS + (ph << 9) + cc,
                     &tileP[(row << 9) + cc]);
    }
    wait_async0();
    __syncthreads();

    for (int kk2 = 0; kk2 < 512; kk2 += 32) {
      int kk = (ph << 9) + kk2;
      v16bf a0 = frag_a(tileP, 512, 0, kk2, lane);
      v16bf a1 = frag_a(tileP, 512, 16, kk2, lane);
      __builtin_prefetch(vt + (size_t)((w << 6) + (lane & 15)) * SS + kk + 256, 0, 1);
#pragma unroll
      for (int t = 0; t < 4; ++t) {
        int n0 = ((w << 2) + t) << 4;
        v16bf b = frag_b(vt, SS, n0, kk, lane);
        acc[0][t] = wmma_bf16(a0, b, acc[0][t]);
        acc[1][t] = wmma_bf16(a1, b, acc[1][t]);
      }
    }
  }

#pragma unroll
  for (int t = 0; t < 4; ++t) {
    int col = (((w << 2) + t) << 4) + (lane & 15);
#pragma unroll
    for (int rt = 0; rt < 2; ++rt) {
#pragma unroll
      for (int r = 0; r < 8; ++r) {
        int m = (rt << 4) + r + (hi << 3);
        out[(size_t)(m0 + m) * DD + col] = acc[rt][t][r];
      }
    }
  }
}

// ---------------------------------------------------------------------------
// Host-side launch. Workspace layout (bytes):
//   xb @0 (16MiB) | wb @16MiB (6MiB) | qb @22MiB (16MiB) | kb @38MiB (16MiB)
//   vtb @54MiB (16MiB, transposed) | Pb @70MiB (32MiB)  -> ~102 MiB total
// ---------------------------------------------------------------------------
extern "C" void kernel_launch(void* const* d_in, const int* in_sizes, int n_in,
                              void* d_out, int out_size, void* d_ws, size_t ws_size,
                              hipStream_t stream) {
  (void)in_sizes; (void)n_in; (void)out_size; (void)ws_size;

  const float* x     = (const float*)d_in[0];
  const float* Wq    = (const float*)d_in[1];
  const float* bq    = (const float*)d_in[2];
  const float* Wk    = (const float*)d_in[3];
  const float* bk    = (const float*)d_in[4];
  const float* Wv    = (const float*)d_in[5];
  const float* bv    = (const float*)d_in[6];
  const float* gamma = (const float*)d_in[7];
  const float* beta  = (const float*)d_in[8];
  float* out = (float*)d_out;

  char* ws = (char*)d_ws;
  const size_t XB_BYTES = (size_t)BB * SS * DD * 2;   // 16 MiB
  const size_t W_BYTES  = (size_t)DD * DD * 2;        //  2 MiB each
  bf16_t* xb  = (bf16_t*)(ws);
  bf16_t* wb  = (bf16_t*)(ws + XB_BYTES);
  bf16_t* qb  = (bf16_t*)(ws + XB_BYTES + 3 * W_BYTES);
  bf16_t* kb  = (bf16_t*)(ws + 2 * XB_BYTES + 3 * W_BYTES);
  bf16_t* vtb = (bf16_t*)(ws + 3 * XB_BYTES + 3 * W_BYTES);
  bf16_t* Pb  = (bf16_t*)(ws + 4 * XB_BYTES + 3 * W_BYTES);

  {
    int n = BB * SS * DD;
    f32_to_bf16_kernel<<<(n + 255) / 256, 256, 0, stream>>>(x, xb, n);
    int nw = DD * DD;
    f32_to_bf16_kernel<<<(nw + 255) / 256, 256, 0, stream>>>(Wq, wb, nw);
    f32_to_bf16_kernel<<<(nw + 255) / 256, 256, 0, stream>>>(Wk, wb + (size_t)DD * DD, nw);
    f32_to_bf16_kernel<<<(nw + 255) / 256, 256, 0, stream>>>(Wv, wb + (size_t)2 * DD * DD, nw);
  }

  // 1) q/k/v = LN(x @ W^T + b)        (32-row tiles)
  qkv_ln_kernel<<<dim3(NROWS / 32, 3), 512, 0, stream>>>(
      xb, wb, bq, bk, bv, gamma, beta, qb, kb, vtb);

  // 2) P = softmax(q @ k^T)           (16-row tiles)
  scores_softmax_kernel<<<NROWS / 16, 512, 0, stream>>>(qb, kb, Pb);

  // 3) out = P @ v                    (32-row tiles)
  pv_kernel<<<NROWS / 32, 512, 0, stream>>>(Pb, vtb, out);
}
